// TriangleMultiplication_13675175870791
// MI455X (gfx1250) — matrix operand order repaired and checked by reading, not compiled, not run on hardware
//
#include <hip/hip_runtime.h>
#include <hip/hip_bf16.h>

// ---------------------------------------------------------------------------
// Types / helpers
// ---------------------------------------------------------------------------
typedef __attribute__((ext_vector_type(16))) __bf16 v16bf;
typedef __attribute__((ext_vector_type(8)))  float  v8f;

union Frag16 {              // 8 VGPRs = 16 bf16 operand fragment
    v16bf v;
    unsigned u[8];
    uint4 q[2];
};

__device__ __forceinline__ unsigned short f2bf(float f) {
    unsigned u = __float_as_uint(f);
    unsigned r = u + 0x7FFFu + ((u >> 16) & 1u);   // round-to-nearest-even
    return (unsigned short)(r >> 16);
}
__device__ __forceinline__ float bf2f(unsigned short h) {
    return __uint_as_float(((unsigned)h) << 16);
}
__device__ __forceinline__ float sigmoidf_(float x) {
    return 1.0f / (1.0f + __expf(-x));
}

#define LL   512
#define CC   128
#define NPOS (LL * LL)          // 262144 positions
#define WSZ  (CC * CC)          // 16384 elements per weight matrix

#define WMMA_BF16(a, b, c) \
    __builtin_amdgcn_wmma_f32_16x16x32_bf16(false, (a), false, (b), (short)0, (c), false, false)

// ---------------------------------------------------------------------------
// Kernel 0: convert the 6 weight matrices (wl, wlg, wr, wrg, wfg, wo) to bf16
// ---------------------------------------------------------------------------
__global__ __launch_bounds__(256)
void convert_weights_kernel(const float* wl, const float* wlg, const float* wr,
                            const float* wrg, const float* wfg, const float* wo,
                            unsigned short* wbf) {
    int gid = blockIdx.x * 256 + threadIdx.x;
    if (gid >= 6 * WSZ) return;
    int m = gid >> 14;          // /16384
    int e = gid & (WSZ - 1);
    const float* src;
    switch (m) {
        case 0: src = wl;  break;
        case 1: src = wlg; break;
        case 2: src = wr;  break;
        case 3: src = wrg; break;
        case 4: src = wfg; break;
        default: src = wo; break;
    }
    wbf[gid] = f2bf(src[e]);
}

// ---------------------------------------------------------------------------
// Kernel 1: LayerNorm + 5 projections via WMMA.
//   64 positions per block (256 threads = 8 waves).
//   Writes Lt[c][i][k]   (bf16, row-major K  -> A operand of stage 2)
//          RtT[c][k][j]  (bf16, K-major      -> B operand of stage 2, no transpose!)
//          gate[p][c]    (bf16 pre-sigmoid)
// ---------------------------------------------------------------------------
__global__ __launch_bounds__(256)
void ln_proj_kernel(const float* __restrict__ act, const float* __restrict__ mask,
                    const float* __restrict__ ng, const float* __restrict__ nb,
                    const float* __restrict__ bl, const float* __restrict__ br,
                    const float* __restrict__ blg, const float* __restrict__ brg,
                    const float* __restrict__ bfg,
                    const unsigned short* __restrict__ wbf,
                    unsigned short* __restrict__ Lt, unsigned short* __restrict__ RtT,
                    unsigned short* __restrict__ gate) {
    __shared__ unsigned short lnrow[64][CC];      // 16 KB bf16 LN'd activations
    const int p0  = blockIdx.x * 64;
    const int tid = threadIdx.x;

    // ---- LayerNorm: thread t < 64 handles one row of 128 channels ----
    if (tid < 64) {
        const float* row = act + (size_t)(p0 + tid) * CC;
        float s = 0.f, s2 = 0.f;
        for (int c = 0; c < CC; c += 4) {
            float4 f = *(const float4*)(row + c);
            s  += f.x + f.y + f.z + f.w;
            s2 += f.x * f.x + f.y * f.y + f.z * f.z + f.w * f.w;
        }
        float mu  = s * (1.0f / CC);
        float var = s2 * (1.0f / CC) - mu * mu;
        float inv = rsqrtf(var + 1e-5f);
        for (int c = 0; c < CC; c += 4) {
            float4 f = *(const float4*)(row + c);
            lnrow[tid][c + 0] = f2bf((f.x - mu) * inv * ng[c + 0] + nb[c + 0]);
            lnrow[tid][c + 1] = f2bf((f.y - mu) * inv * ng[c + 1] + nb[c + 1]);
            lnrow[tid][c + 2] = f2bf((f.z - mu) * inv * ng[c + 2] + nb[c + 2]);
            lnrow[tid][c + 3] = f2bf((f.w - mu) * inv * ng[c + 3] + nb[c + 3]);
        }
    }
    __syncthreads();

    const int lane = tid & 31;
    const int wv   = tid >> 5;
    const int m    = lane & 15;
    const int hi   = lane >> 4;

    // 96 jobs: type 0 = left(wl,wlg), 1 = right(wr,wrg), 2 = gate(wfg,wfg)
    for (int job = wv; job < 96; job += 8) {
        const int type = job / 32;
        const int rr   = job % 32;
        const int mi   = rr & 3;          // 16-row subtile
        const int n0   = (rr >> 2) * 16;  // 16-col subtile (of 128)

        const int m0i = (type == 0) ? 0 : (type == 1) ? 2 : 4;
        const int m1i = (type == 0) ? 1 : (type == 1) ? 3 : 4;   // type 2 pairs with itself
        const unsigned short* w0 = wbf + m0i * WSZ;
        const unsigned short* w1 = wbf + m1i * WSZ;

        // ---- preload all fragments (lets the scheduler hoist loads) ----
        Frag16 a[4], b0[4], b1[4];
#pragma unroll
        for (int kt = 0; kt < 4; ++kt) {
            const int kb  = kt * 32;
            const int row = mi * 16 + m;
            a[kt].q[0] = *(const uint4*)&lnrow[row][kb + 8 * hi];
            a[kt].q[1] = *(const uint4*)&lnrow[row][kb + 16 + 8 * hi];
            const int ncol = n0 + m;                    // B: lane = N, elements = K (table order)
#pragma unroll
            for (int t = 0; t < 8; ++t) {
                const int k0 = kb + 8 * hi + 2 * (t & 3) + 16 * (t >> 2);
                b0[kt].u[t] = (unsigned)w0[k0 * CC + ncol] | ((unsigned)w0[(k0 + 1) * CC + ncol] << 16);
                b1[kt].u[t] = (unsigned)w1[k0 * CC + ncol] | ((unsigned)w1[(k0 + 1) * CC + ncol] << 16);
            }
        }
        v8f acc0 = {}, acc1 = {};
#pragma unroll
        for (int kt = 0; kt < 4; ++kt) {
            acc0 = WMMA_BF16(a[kt].v, b0[kt].v, acc0);
            acc1 = WMMA_BF16(a[kt].v, b1[kt].v, acc1);
        }

        // ---- epilogue / scatter ----
        const int c = n0 + (lane & 15);
#pragma unroll
        for (int r8 = 0; r8 < 8; ++r8) {
            const int mm = r8 + 8 * hi;
            const int p  = p0 + mi * 16 + mm;
            if (type == 2) {
                gate[(size_t)p * CC + c] = f2bf(acc0[r8] + bfg[c]);
            } else {
                const int i  = p >> 9;       // row index of position
                const int kp = p & 511;      // col index of position
                const float pm = mask[i] * mask[kp];
                const float bb = (type == 0) ? bl[c]  : br[c];
                const float bg = (type == 0) ? blg[c] : brg[c];
                const float val = pm * (acc0[r8] + bb) * sigmoidf_(acc1[r8] + bg);
                if (type == 0)
                    Lt [(size_t)c * NPOS + (size_t)i  * LL + kp] = f2bf(val);
                else   // right[j,k,c] stored K-major: RtT[c][k][j]
                    RtT[(size_t)c * NPOS + (size_t)kp * LL + i ] = f2bf(val);
            }
        }
    }
}

// ---------------------------------------------------------------------------
// Kernel 2: triangle einsum out[i,j,c] = sum_k Lt[c][i][k] * RtT[c][k][j]
//   128 batched 512x512x512 GEMMs (bf16 WMMA, fp32 accum), operands L2-resident.
//   Block = one channel c, one 64x64 output tile. 8 waves x 2 WMMA tiles.
//   Software pipelined: frags for step kt+1 load while step kt computes.
// ---------------------------------------------------------------------------
__global__ __launch_bounds__(256)
void triangle_kernel(const unsigned short* __restrict__ Lt,
                     const unsigned short* __restrict__ RtT,
                     unsigned short* __restrict__ outpre) {
    const int c  = blockIdx.z;
    const int i0 = blockIdx.x * 64;
    const int j0 = blockIdx.y * 64;
    const unsigned short* Lc = Lt  + (size_t)c * NPOS;
    const unsigned short* Rc = RtT + (size_t)c * NPOS;

    const int tid  = threadIdx.x;
    const int lane = tid & 31;
    const int wv   = tid >> 5;
    const int mi   = wv & 3;          // 16-row subtile of 64
    const int njG  = wv >> 2;         // 0..1 -> 32-col group
    const int m    = lane & 15;
    const int hi   = lane >> 4;

    const unsigned short* arow = Lc + (size_t)(i0 + mi * 16 + m) * LL;
    const int jb0 = j0 + njG * 32;    // first 16-col subtile base
    const int jb1 = jb0 + 16;         // second

    Frag16 aC, b0C, b1C, aN, b0N, b1N;
    // prologue: load kt = 0
    aC.q[0]  = *(const uint4*)&arow[8 * hi];
    aC.q[1]  = *(const uint4*)&arow[16 + 8 * hi];
#pragma unroll
    for (int t = 0; t < 8; ++t) {
        const size_t k0 = 8 * hi + 2 * (t & 3) + 16 * (t >> 2);
        b0C.u[t] = (unsigned)Rc[k0 * LL + jb0 + m] | ((unsigned)Rc[(k0 + 1) * LL + jb0 + m] << 16);
        b1C.u[t] = (unsigned)Rc[k0 * LL + jb1 + m] | ((unsigned)Rc[(k0 + 1) * LL + jb1 + m] << 16);
    }

    v8f acc0 = {}, acc1 = {};
#pragma unroll
    for (int kt = 0; kt < 15; ++kt) {
        const int kb = (kt + 1) * 32;
        // prefetch two slabs ahead (global_prefetch_b8)
        const int kpf = (kb + 32 <= 480) ? kb + 32 : 480;
        __builtin_prefetch(&arow[kpf], 0, 1);
        __builtin_prefetch(&Rc[(size_t)(kpf + lane) * LL + jb0], 0, 1);
        // load next K step
        aN.q[0]  = *(const uint4*)&arow[kb + 8 * hi];
        aN.q[1]  = *(const uint4*)&arow[kb + 16 + 8 * hi];
#pragma unroll
        for (int t = 0; t < 8; ++t) {
            const size_t k0 = kb + 8 * hi + 2 * (t & 3) + 16 * (t >> 2);
            b0N.u[t] = (unsigned)Rc[k0 * LL + jb0 + m] | ((unsigned)Rc[(k0 + 1) * LL + jb0 + m] << 16);
            b1N.u[t] = (unsigned)Rc[k0 * LL + jb1 + m] | ((unsigned)Rc[(k0 + 1) * LL + jb1 + m] << 16);
        }
        // compute current
        acc0 = WMMA_BF16(aC.v, b0C.v, acc0);
        acc1 = WMMA_BF16(aC.v, b1C.v, acc1);
        aC = aN; b0C = b0N; b1C = b1N;
    }
    acc0 = WMMA_BF16(aC.v, b0C.v, acc0);
    acc1 = WMMA_BF16(aC.v, b1C.v, acc1);

    // ---- scatter bf16 out_pre[(i*512+j)*128 + c] ----
#pragma unroll
    for (int r8 = 0; r8 < 8; ++r8) {
        const int ii  = i0 + mi * 16 + r8 + 8 * hi;
        const int jj0 = jb0 + (lane & 15);
        const int jj1 = jb1 + (lane & 15);
        outpre[((size_t)ii * LL + jj0) * CC + c] = f2bf(acc0[r8]);
        outpre[((size_t)ii * LL + jj1) * CC + c] = f2bf(acc1[r8]);
    }
}

// ---------------------------------------------------------------------------
// Kernel 3: LN(out_pre) @ wo + bo, gated by sigmoid(gate), fp32 output.
// ---------------------------------------------------------------------------
__global__ __launch_bounds__(256)
void out_kernel(const unsigned short* __restrict__ outpre,
                const unsigned short* __restrict__ gate,
                const unsigned short* __restrict__ wo_bf,
                const float* __restrict__ fng, const float* __restrict__ fnb,
                const float* __restrict__ bo, float* __restrict__ out) {
    __shared__ unsigned short lnrow[64][CC];
    const int p0  = blockIdx.x * 64;
    const int tid = threadIdx.x;

    if (tid < 64) {
        const unsigned short* row = outpre + (size_t)(p0 + tid) * CC;
        float s = 0.f, s2 = 0.f;
        for (int c = 0; c < CC; c += 2) {
            unsigned u = *(const unsigned*)(row + c);
            float f0 = bf2f((unsigned short)(u & 0xFFFF));
            float f1 = bf2f((unsigned short)(u >> 16));
            s += f0 + f1; s2 += f0 * f0 + f1 * f1;
        }
        float mu  = s * (1.0f / CC);
        float var = s2 * (1.0f / CC) - mu * mu;
        float inv = rsqrtf(var + 1e-5f);
        for (int c = 0; c < CC; c += 2) {
            unsigned u = *(const unsigned*)(row + c);
            float f0 = bf2f((unsigned short)(u & 0xFFFF));
            float f1 = bf2f((unsigned short)(u >> 16));
            lnrow[tid][c + 0] = f2bf((f0 - mu) * inv * fng[c + 0] + fnb[c + 0]);
            lnrow[tid][c + 1] = f2bf((f1 - mu) * inv * fng[c + 1] + fnb[c + 1]);
        }
    }
    __syncthreads();

    const int lane = tid & 31;
    const int wv   = tid >> 5;
    const int m    = lane & 15;
    const int hi   = lane >> 4;

    for (int job = wv; job < 32; job += 8) {
        const int mi = job & 3;
        const int n0 = (job >> 2) * 16;

        Frag16 a[4], b[4];
#pragma unroll
        for (int kt = 0; kt < 4; ++kt) {
            const int kb  = kt * 32;
            const int row = mi * 16 + m;
            a[kt].q[0] = *(const uint4*)&lnrow[row][kb + 8 * hi];
            a[kt].q[1] = *(const uint4*)&lnrow[row][kb + 16 + 8 * hi];
#pragma unroll
            for (int t = 0; t < 8; ++t) {
                const int k0 = kb + 8 * hi + 2 * (t & 3) + 16 * (t >> 2);
                b[kt].u[t] = (unsigned)wo_bf[k0 * CC + n0 + m] | ((unsigned)wo_bf[(k0 + 1) * CC + n0 + m] << 16);
            }
        }
        v8f acc = {};
#pragma unroll
        for (int kt = 0; kt < 4; ++kt)
            acc = WMMA_BF16(a[kt].v, b[kt].v, acc);

        const int c = n0 + (lane & 15);
#pragma unroll
        for (int r8 = 0; r8 < 8; ++r8) {
            const int p = p0 + mi * 16 + r8 + 8 * hi;
            const float g = sigmoidf_(bf2f(gate[(size_t)p * CC + c]));
            out[(size_t)p * CC + c] = (acc[r8] + bo[c]) * g;
        }
    }
}

// ---------------------------------------------------------------------------
// Launch
// ---------------------------------------------------------------------------
extern "C" void kernel_launch(void* const* d_in, const int* in_sizes, int n_in,
                              void* d_out, int out_size, void* d_ws, size_t ws_size,
                              hipStream_t stream) {
    const float* act   = (const float*)d_in[0];
    const float* mask  = (const float*)d_in[1];
    const float* ng    = (const float*)d_in[2];
    const float* nb    = (const float*)d_in[3];
    const float* wl    = (const float*)d_in[4];
    const float* bl    = (const float*)d_in[5];
    const float* wr    = (const float*)d_in[6];
    const float* br    = (const float*)d_in[7];
    const float* wlg   = (const float*)d_in[8];
    const float* blg   = (const float*)d_in[9];
    const float* wrg   = (const float*)d_in[10];
    const float* brg   = (const float*)d_in[11];
    const float* fng   = (const float*)d_in[12];
    const float* fnb   = (const float*)d_in[13];
    const float* wo    = (const float*)d_in[14];
    const float* bo    = (const float*)d_in[15];
    const float* wfg   = (const float*)d_in[16];
    const float* bfg   = (const float*)d_in[17];

    // Workspace layout (bf16):
    //   weights 6*16384          = 196608 B
    //   Lt  [c][i][k]            = 67108864 B
    //   RtT [c][k][j]            = 67108864 B
    //   gate[p][c]               = 67108864 B
    //   out_pre[p][c]            = 67108864 B   (total ~268.6 MB)
    char* ws = (char*)d_ws;
    unsigned short* wbf    = (unsigned short*)ws;
    unsigned short* Lt     = (unsigned short*)(ws + 196608);
    unsigned short* RtT    = Lt + (size_t)CC * NPOS;
    unsigned short* gatep  = RtT + (size_t)CC * NPOS;
    unsigned short* outpre = gatep + (size_t)CC * NPOS;

    convert_weights_kernel<<<(6 * WSZ + 255) / 256, 256, 0, stream>>>(
        wl, wlg, wr, wrg, wfg, wo, wbf);

    ln_proj_kernel<<<NPOS / 64, 256, 0, stream>>>(
        act, mask, ng, nb, bl, br, blg, brg, bfg, wbf, Lt, RtT, gatep);

    triangle_kernel<<<dim3(LL / 64, LL / 64, CC), 256, 0, stream>>>(
        Lt, RtT, outpre);

    out_kernel<<<NPOS / 64, 256, 0, stream>>>(
        outpre, gatep, wbf + 5 * WSZ, fng, fnb, bo, (float*)d_out);
}
